// CausalSelfAttention_32555852104300
// MI455X (gfx1250) — compile-verified, matrix operand order checked by reading, not run on hardware
//
#include <hip/hip_runtime.h>

// ---------------------------------------------------------------------------
// CDNA5 (gfx1250) wave32 WMMA types
// ---------------------------------------------------------------------------
typedef __attribute__((ext_vector_type(16))) __bf16 v16bf;
typedef __attribute__((ext_vector_type(8)))  float  v8f;
typedef __attribute__((ext_vector_type(4)))  unsigned int u32x4;
typedef __attribute__((ext_vector_type(8)))  int          i32x8;
typedef __attribute__((ext_vector_type(4)))  int          i32x4;

#if defined(__has_builtin)
#if __has_builtin(__builtin_amdgcn_tensor_load_to_lds) && \
    __has_builtin(__builtin_amdgcn_s_wait_tensorcnt)
#define USE_TDM 1
#endif
#endif

union FragB {
    v16bf v;
    uint4 q[2];
    unsigned short u[16];
};

__device__ __forceinline__ unsigned short f2bf(float f) {
    union { __bf16 b; unsigned short u; } cv;
    cv.b = (__bf16)f;
    return cv.u;
}

__device__ __forceinline__ v8f wmma_bf16(v16bf a, v16bf b, v8f c) {
    return __builtin_amdgcn_wmma_f32_16x16x32_bf16(
        false, a, false, b, (short)0, c, false, false);
}

#ifdef USE_TDM
// Issue a TDM 2-D tile DMA: tile_d1 rows x tile_d0 elements (2-byte) from a
// row-major tensor with row stride `stride0` (elements) into LDS at byte
// offset `lds_byte`.  D# packing per CDNA5 ISA.
// This toolchain exposes the 6-arg builtin:
//   (u32x4 g0, i32x8 g1, i32x4 g2, i32x4 g3, i32x8 pad, i32 cpol)
__device__ __forceinline__ void tdm_load_2d_bf16(unsigned lds_byte, const void* gaddr,
                                                 unsigned tensor_d0, unsigned tensor_d1,
                                                 unsigned stride0,
                                                 unsigned tile_d0, unsigned tile_d1) {
    const unsigned long long ga = (unsigned long long)gaddr;
    u32x4 g0;
    g0[0] = 1u;                                            // count=1, user descriptor
    g0[1] = lds_byte;                                      // lds_addr (bytes)
    g0[2] = (unsigned)ga;                                  // global_addr[31:0]
    g0[3] = (unsigned)((ga >> 32) & 0x01FFFFFFu) | (2u << 30);  // addr[56:32], type=2
    i32x8 g1;
    g1[0] = (int)(1u << 16);                               // data_size=1 (2 bytes)
    g1[1] = (int)((tensor_d0 & 0xFFFFu) << 16);            // tensor_dim0[15:0]
    g1[2] = (int)((tensor_d0 >> 16) | ((tensor_d1 & 0xFFFFu) << 16));
    g1[3] = (int)((tensor_d1 >> 16) | (tile_d0 << 16));    // tile_dim0
    g1[4] = (int)tile_d1;                                  // tile_dim1 (tile_dim2=0)
    g1[5] = (int)stride0;                                  // tensor_dim0_stride[31:0]
    g1[6] = 0;
    g1[7] = 0;
    const i32x4 gz4 = {0, 0, 0, 0};                        // <=2-D: groups 2/3 unused
    const i32x8 gz8 = {0, 0, 0, 0, 0, 0, 0, 0};
    __builtin_amdgcn_tensor_load_to_lds(g0, g1, gz4, gz4, gz8, 0);
}
#endif

// ---------------------------------------------------------------------------
// fp32 -> bf16 bulk convert (8 elements / thread)
// ---------------------------------------------------------------------------
__global__ __launch_bounds__(256)
void cvt_f32_to_bf16(const float* __restrict__ X, unsigned short* __restrict__ Xb) {
    const size_t i = (size_t)(blockIdx.x * 256 + threadIdx.x) * 8;
    const float4 f0 = *(const float4*)(X + i);
    const float4 f1 = *(const float4*)(X + i + 4);
    union { uint4 q; unsigned short u[8]; } o;
    o.u[0] = f2bf(f0.x); o.u[1] = f2bf(f0.y); o.u[2] = f2bf(f0.z); o.u[3] = f2bf(f0.w);
    o.u[4] = f2bf(f1.x); o.u[5] = f2bf(f1.y); o.u[6] = f2bf(f1.z); o.u[7] = f2bf(f1.w);
    *(uint4*)(Xb + i) = o.q;
}

// ---------------------------------------------------------------------------
// fp32 (K x N) -> bf16 transposed (N x K)
// ---------------------------------------------------------------------------
__global__ __launch_bounds__(256)
void transpose_to_bf16(const float* __restrict__ W, unsigned short* __restrict__ Wt,
                       int Kd, int N) {
    __shared__ float tile[32][33];
    const int tx = threadIdx.x, ty = threadIdx.y;
    const int gn = blockIdx.x * 32, gk = blockIdx.y * 32;
#pragma unroll
    for (int i = 0; i < 4; i++) {
        int r = ty + i * 8;
        tile[r][tx] = W[(size_t)(gk + r) * N + gn + tx];
    }
    __syncthreads();
#pragma unroll
    for (int i = 0; i < 4; i++) {
        int r = ty + i * 8;
        Wt[(size_t)(gn + r) * Kd + gk + tx] = f2bf(tile[tx][r]);
    }
}

// ---------------------------------------------------------------------------
// GEMM  C[M,N](f32) = A[M,K](bf16) * Bt[N,K](bf16)^T + bias
// Block = 256 threads (8 waves); block tile 128x64; wave tile 32x32 (2x2 WMMA).
// B tiles (shared by all 8 waves) are staged into LDS by the Tensor Data Mover
// (double-buffered, TENSORcnt-tracked) so DMA of chunk n+1 overlaps the WMMAs
// of chunk n.  A fragments are direct global_load_b128 (only 2-way shared,
// L2-resident).
// ---------------------------------------------------------------------------
__global__ __launch_bounds__(256)
void gemm_bf16(const unsigned short* __restrict__ A, const unsigned short* __restrict__ Bt,
               const float* __restrict__ bias, float* __restrict__ Co,
               int M, int N, int Kd) {
    extern __shared__ unsigned short bt[];  // 2 buffers x 64 rows x 64 halves (16 KB)

    const int tid = threadIdx.x;
    const int w  = tid >> 5;
    const int l  = tid & 31;
    const int ln = l & 15;
    const int lg = l >> 4;
    const int wm = w & 3;    // 4 waves along M
    const int wn = w >> 2;   // 2 waves along N
    const int row0 = blockIdx.y * 128 + wm * 32;
    const int colBlk = blockIdx.x * 64;

    const v8f z = {0.f, 0.f, 0.f, 0.f, 0.f, 0.f, 0.f, 0.f};
    v8f acc[2][2] = {{z, z}, {z, z}};

#ifdef USE_TDM
    if (w == 0)
        tdm_load_2d_bf16(0, Bt + (size_t)colBlk * Kd, Kd, N, Kd, 64, 64);
#endif

    for (int kc = 0; kc < Kd; kc += 64) {
        const int buf = (kc >> 6) & 1;
        unsigned short* bbase = bt + buf * 4096;

#ifdef USE_TDM
        if (w == 0) __builtin_amdgcn_s_wait_tensorcnt((short)0);
        __syncthreads();
        if (w == 0 && kc + 64 < Kd)
            tdm_load_2d_bf16((unsigned)((buf ^ 1) * 8192),
                             Bt + (size_t)colBlk * Kd + kc + 64, Kd, N, Kd, 64, 64);
#else
        __syncthreads();
        {   // cooperative stage: 256 threads x 16 halves = 64x64 tile
            const int trow = (tid * 16) >> 6;
            const int tcol = (tid * 16) & 63;
            const unsigned short* src = Bt + (size_t)(colBlk + trow) * Kd + kc + tcol;
            *(uint4*)(bbase + trow * 64 + tcol)     = *(const uint4*)(src);
            *(uint4*)(bbase + trow * 64 + tcol + 8) = *(const uint4*)(src + 8);
        }
        __syncthreads();
#endif
        // prefetch A two chunks ahead into GL2
        if (kc + 128 < Kd)
            __builtin_prefetch(A + (size_t)(row0 + ln) * Kd + kc + 128, 0, 1);

#pragma unroll
        for (int ks = 0; ks < 64; ks += 32) {
            FragB a[2], b[2];
#pragma unroll
            for (int mi = 0; mi < 2; mi++) {
                const unsigned short* ap =
                    A + (size_t)(row0 + mi * 16 + ln) * Kd + kc + ks + lg * 8;
                a[mi].q[0] = *(const uint4*)(ap);
                a[mi].q[1] = *(const uint4*)(ap + 16);
            }
#pragma unroll
            for (int ni = 0; ni < 2; ni++) {
                const unsigned short* bp =
                    bbase + (wn * 32 + ni * 16 + ln) * 64 + ks + lg * 8;
                b[ni].q[0] = *(const uint4*)(bp);
                b[ni].q[1] = *(const uint4*)(bp + 16);
            }
#pragma unroll
            for (int mi = 0; mi < 2; mi++)
#pragma unroll
                for (int ni = 0; ni < 2; ni++)
                    acc[mi][ni] = wmma_bf16(a[mi].v, b[ni].v, acc[mi][ni]);
        }
    }

#pragma unroll
    for (int mi = 0; mi < 2; mi++)
#pragma unroll
        for (int ni = 0; ni < 2; ni++) {
            const int col = colBlk + wn * 32 + ni * 16 + ln;
            const float bv = bias[col];
#pragma unroll
            for (int r = 0; r < 8; r++) {
                const int row = row0 + mi * 16 + r + 8 * lg;
                Co[(size_t)row * N + col] = acc[mi][ni][r] + bv;
            }
        }
}

// ---------------------------------------------------------------------------
// RoPE + repack qkv(f32) -> Q,K bf16 [BH][T][64], V^T bf16 [BH][64][T]
// ---------------------------------------------------------------------------
__global__ __launch_bounds__(256)
void rope_pack(const float* __restrict__ qkv, unsigned short* __restrict__ Qb,
               unsigned short* __restrict__ Kb, unsigned short* __restrict__ Vt) {
    const int tid = blockIdx.x * 256 + threadIdx.x;  // B*H*T*32 threads
    const int d  = tid & 31;
    const int t  = (tid >> 5) & 2047;
    const int bh = tid >> 16;  // 0..63
    const int b  = bh >> 4;
    const size_t base = ((size_t)(b * 2048 + t)) * 3072 + (size_t)(bh & 15) * 64 + d;

    const float q1 = qkv[base],        q2 = qkv[base + 32];
    const float k1 = qkv[base + 1024], k2 = qkv[base + 1024 + 32];
    const float v1 = qkv[base + 2048], v2 = qkv[base + 2048 + 32];

    const float freq  = __expf((float)(d & 15) * (-9.210340371976184f / 16.0f));
    const float theta = (float)t * freq;
    float s, c;
    sincosf(theta, &s, &c);

    const size_t qkidx = ((size_t)bh * 2048 + t) * 64 + d;
    Qb[qkidx]      = f2bf(q1 * c - q2 * s);
    Qb[qkidx + 32] = f2bf(q2 * c + q1 * s);
    Kb[qkidx]      = f2bf(k1 * c - k2 * s);
    Kb[qkidx + 32] = f2bf(k2 * c + k1 * s);

    Vt[((size_t)bh * 64 + d) * 2048 + t]      = f2bf(v1);
    Vt[((size_t)bh * 64 + d + 32) * 2048 + t] = f2bf(v2);
}

// ---------------------------------------------------------------------------
// Flash attention.  Block = 128 threads (4 waves), 64 queries/block,
// 16 queries/wave, 32-key blocks, online softmax fp32, GEMMs via WMMA bf16.
// ---------------------------------------------------------------------------
__global__ __launch_bounds__(128)
void attn_fwd(const unsigned short* __restrict__ Q, const unsigned short* __restrict__ K,
              const unsigned short* __restrict__ Vt, unsigned short* __restrict__ Yb) {
    constexpr int T = 2048;
    __shared__ __align__(16) unsigned short plds[4][16 * 40];  // P transpose buffers

    const int w  = threadIdx.x >> 5;
    const int l  = threadIdx.x & 31;
    const int ln = l & 15;
    const int lg = l >> 4;
    const int bh = blockIdx.y;
    const int qbase = blockIdx.x * 64 + w * 16;

    const unsigned short* Qp = Q  + (size_t)bh * T * 64;
    const unsigned short* Kp = K  + (size_t)bh * T * 64;
    const unsigned short* Vp = Vt + (size_t)bh * 64 * T;

    FragB qa[2];
    {
        const unsigned short* qp = Qp + (size_t)(qbase + ln) * 64;
        qa[0].q[0] = *(const uint4*)(qp + lg * 8);
        qa[0].q[1] = *(const uint4*)(qp + 16 + lg * 8);
        qa[1].q[0] = *(const uint4*)(qp + 32 + lg * 8);
        qa[1].q[1] = *(const uint4*)(qp + 48 + lg * 8);
    }

    const v8f z = {0.f, 0.f, 0.f, 0.f, 0.f, 0.f, 0.f, 0.f};
    v8f o[4] = {z, z, z, z};
    float mrow[8], lrow[8];
#pragma unroll
    for (int r = 0; r < 8; r++) { mrow[r] = -__builtin_inff(); lrow[r] = 0.f; }

    for (int kb = 0; kb < qbase + 16; kb += 32) {
        // ---- S = Q * K^T  (16 x 32) ----
        v8f s[2] = {z, z};
#pragma unroll
        for (int nt = 0; nt < 2; nt++) {
            const unsigned short* kp = Kp + (size_t)(kb + nt * 16 + ln) * 64 + lg * 8;
            FragB kf;
            kf.q[0] = *(const uint4*)(kp);
            kf.q[1] = *(const uint4*)(kp + 16);
            s[nt] = wmma_bf16(qa[0].v, kf.v, s[nt]);
            kf.q[0] = *(const uint4*)(kp + 32);
            kf.q[1] = *(const uint4*)(kp + 48);
            s[nt] = wmma_bf16(qa[1].v, kf.v, s[nt]);
        }

        // ---- scale, causal mask, online softmax ----
        float al[8];
#pragma unroll
        for (int r = 0; r < 8; r++) {
            const int qrow = qbase + r + 8 * lg;
            float v0 = s[0][r] * 0.125f;
            float v1 = s[1][r] * 0.125f;
            if (kb + ln > qrow)      v0 = -3.0e38f;
            if (kb + 16 + ln > qrow) v1 = -3.0e38f;
            float lm = fmaxf(v0, v1);
            lm = fmaxf(lm, __shfl_xor(lm, 1));
            lm = fmaxf(lm, __shfl_xor(lm, 2));
            lm = fmaxf(lm, __shfl_xor(lm, 4));
            lm = fmaxf(lm, __shfl_xor(lm, 8));
            const float mn = fmaxf(mrow[r], lm);
            al[r] = __expf(mrow[r] - mn);
            const float p0 = __expf(v0 - mn);
            const float p1 = __expf(v1 - mn);
            s[0][r] = p0;
            s[1][r] = p1;
            float rs = p0 + p1;
            rs += __shfl_xor(rs, 1);
            rs += __shfl_xor(rs, 2);
            rs += __shfl_xor(rs, 4);
            rs += __shfl_xor(rs, 8);
            lrow[r] = lrow[r] * al[r] + rs;
            mrow[r] = mn;
        }
#pragma unroll
        for (int t = 0; t < 4; t++)
#pragma unroll
            for (int r = 0; r < 8; r++) o[t][r] *= al[r];

        // ---- P: C-layout -> A-layout via wave-local LDS ----
        unsigned short* pl = &plds[w][0];
#pragma unroll
        for (int r = 0; r < 8; r++) {
            pl[(r + 8 * lg) * 40 + ln]      = f2bf(s[0][r]);
            pl[(r + 8 * lg) * 40 + 16 + ln] = f2bf(s[1][r]);
        }
        asm volatile("s_wait_dscnt 0" ::: "memory");
        FragB pf;
        pf.q[0] = *(const uint4*)(pl + ln * 40 + lg * 8);
        pf.q[1] = *(const uint4*)(pl + ln * 40 + 16 + lg * 8);

        // ---- O += P * V ----
#pragma unroll
        for (int dt = 0; dt < 4; dt++) {
            const unsigned short* vp = Vp + (size_t)(dt * 16 + ln) * T + kb + lg * 8;
            FragB vf;
            vf.q[0] = *(const uint4*)(vp);
            vf.q[1] = *(const uint4*)(vp + 16);
            o[dt] = wmma_bf16(pf.v, vf.v, o[dt]);
        }
    }

    // ---- normalize & write y[B][T][C] as bf16 for the projection GEMM ----
    const int b = bh >> 4, h = bh & 15;
#pragma unroll
    for (int r = 0; r < 8; r++) {
        const float inv = 1.0f / lrow[r];
        const int trow = qbase + r + 8 * lg;
        unsigned short* yp = Yb + ((size_t)(b * T + trow)) * 1024 + h * 64 + ln;
#pragma unroll
        for (int dt = 0; dt < 4; dt++) yp[dt * 16] = f2bf(o[dt][r] * inv);
    }
}

// ---------------------------------------------------------------------------
// Host launcher
// ---------------------------------------------------------------------------
extern "C" void kernel_launch(void* const* d_in, const int* in_sizes, int n_in,
                              void* d_out, int out_size, void* d_ws, size_t ws_size,
                              hipStream_t stream) {
    const float* x      = (const float*)d_in[0];
    const float* W_attn = (const float*)d_in[1];
    const float* b_attn = (const float*)d_in[2];
    const float* W_proj = (const float*)d_in[3];
    const float* b_proj = (const float*)d_in[4];

    constexpr size_t M = 8192;  // B*T
    constexpr size_t OFF_QKV = 0;                          // 96 MB f32 (later: yb)
    constexpr size_t OFF_Q   = M * 3072 * 4;               // Q bf16 (16 MB)
    constexpr size_t OFF_K   = OFF_Q + M * 64 * 16 * 2;    // K bf16
    constexpr size_t OFF_V   = OFF_K + M * 64 * 16 * 2;    // V^T bf16
    constexpr size_t OFF_WA  = OFF_V + M * 64 * 16 * 2;    // Wt_attn bf16 (6 MB)
    constexpr size_t OFF_WP  = OFF_WA + 3072 * 1024 * 2;   // Wt_proj bf16 (2 MB)
    constexpr size_t OFF_XB  = OFF_WP + 1024 * 1024 * 2;   // x bf16 (16 MB)

    char* ws = (char*)d_ws;
    float*          qkv = (float*)(ws + OFF_QKV);
    unsigned short* Qb  = (unsigned short*)(ws + OFF_Q);
    unsigned short* Kb  = (unsigned short*)(ws + OFF_K);
    unsigned short* Vt  = (unsigned short*)(ws + OFF_V);
    unsigned short* WtA = (unsigned short*)(ws + OFF_WA);
    unsigned short* WtP = (unsigned short*)(ws + OFF_WP);
    unsigned short* xb  = (unsigned short*)(ws + OFF_XB);
    unsigned short* yb  = (unsigned short*)(ws + OFF_QKV);  // reuse qkv region

    // 0) precision/layout prep
    cvt_f32_to_bf16<<<dim3((M * 1024) / (256 * 8)), 256, 0, stream>>>(x, xb);
    transpose_to_bf16<<<dim3(96, 32), dim3(32, 8), 0, stream>>>(W_attn, WtA, 1024, 3072);
    transpose_to_bf16<<<dim3(32, 32), dim3(32, 8), 0, stream>>>(W_proj, WtP, 1024, 1024);

    // 1) qkv = x @ W_attn + b_attn   (TDM-staged B tiles, 16 KB dynamic LDS)
    gemm_bf16<<<dim3(48, 64), 256, 16384, stream>>>(xb, WtA, b_attn, qkv,
                                                    8192, 3072, 1024);

    // 2) rope + pack to per-head bf16 layouts
    rope_pack<<<16384, 256, 0, stream>>>(qkv, Qb, Kb, Vt);

    // 3) flash attention -> yb (bf16, [B][T][C]); overwrites qkv region
    attn_fwd<<<dim3(32, 64), 128, 0, stream>>>(Qb, Kb, Vt, yb);

    // 4) out = y @ W_proj + b_proj
    gemm_bf16<<<dim3(16, 64), 256, 16384, stream>>>(yb, WtP, b_proj, (float*)d_out,
                                                    8192, 1024, 1024);
}
